// CAML_model_40501541601657
// MI455X (gfx1250) — matrix operand order repaired and checked by reading, not compiled, not run on hardware
//
#include <hip/hip_runtime.h>
#include <hip/hip_bf16.h>
#include <math.h>

// ---------------- problem constants ----------------
constexpr int B    = 8;
constexpr int L    = 2500;
constexpr int EMB  = 100;
constexpr int NF   = 50;
constexpr int FW   = 9;
constexpr int PAD  = 4;
constexpr int NL   = 8921;

constexpr int NFP  = 64;             // NF padded to K-multiple for bf16 WMMA (2x K=32)
constexpr int LP   = 2528;           // L padded to multiple of 32 (79 * 32, 158 * 16)
constexpr int NLT  = (NL + 15) / 16; // 558 label tiles of 16

constexpr int OUT_YHAT  = 0;             // [B*NL] floats
constexpr int OUT_LOSS  = B * NL;        // 1 float
constexpr int OUT_ALPHA = B * NL + 1;    // [B*NL*L] floats

typedef __attribute__((ext_vector_type(16))) __bf16 v16bf;
typedef __attribute__((ext_vector_type(8)))  float  v8f;

// =====================================================================
// Kernel 1: embedding gather + 1D conv (pad 4) + tanh
//   -> Hpad [B][LP][NFP] bf16 (position-major, for scores GEMM B-frags)
//   -> HT   [B][NFP][LP] bf16 (feature-major,  for context GEMM B-frags)
// Zero-padded in f (>=NF) and pos (>=L).
// Embedding rows are gathered straight into LDS with the CDNA5 async
// copy path (GLOBAL_LOAD_ASYNC_TO_LDS_B128, ASYNCcnt-tracked).
// =====================================================================
__global__ __launch_bounds__(256)
void caml_embed_conv_tanh(const int* __restrict__ x,
                          const float* __restrict__ emb,
                          const float* __restrict__ conv_w,
                          const float* __restrict__ conv_b,
                          __bf16* __restrict__ Hpad,
                          __bf16* __restrict__ HT) {
  constexpr int ROWS = 64 + 2 * PAD;               // 72 staged rows
  constexpr int CHPR = (EMB * 4) / 16;             // 25 x 16B chunks per row
  alignas(16) __shared__ float sxe[ROWS][EMB];     // 28.8 KB

  const int b   = blockIdx.y;
  const int l0  = blockIdx.x * 64;
  const int tid = threadIdx.x;

  // stage embedding rows [l0-4, l0+67] into LDS: one 16B chunk per lane-iter
  for (int c = tid; c < ROWS * CHPR; c += 256) {
    const int r = c / CHPR;
    const int q = c % CHPR;               // chunk within row
    const int p = l0 + r - PAD;
    if (p >= 0 && p < L) {
      const int tok = x[b * L + p];
      const float* g = emb + (size_t)tok * EMB + q * 4;
      // LDS byte address = low 32 bits of the generic pointer (ISA 10.2)
      const unsigned lds = (unsigned)(size_t)(&sxe[r][q * 4]);
      asm volatile("global_load_async_to_lds_b128 %0, %1, off"
                   :: "v"(lds), "v"(g)
                   : "memory");
    } else {
      sxe[r][q * 4 + 0] = 0.0f;
      sxe[r][q * 4 + 1] = 0.0f;
      sxe[r][q * 4 + 2] = 0.0f;
      sxe[r][q * 4 + 3] = 0.0f;
    }
  }
  asm volatile("s_wait_asynccnt 0" ::: "memory");   // my async copies done
  __syncthreads();                                  // everyone's done

  // each thread produces several (pos, f) outputs
  for (int o = tid; o < 64 * NFP; o += 256) {
    const int pos = o / NFP;
    const int f   = o % NFP;
    const int gp  = l0 + pos;
    if (gp >= LP) continue;
    float outv = 0.0f;
    if (f < NF && gp < L) {
      float acc = conv_b[f];
      const float* wf = conv_w + (size_t)f * EMB * FW;
      for (int e = 0; e < EMB; ++e) {
        const float* we = wf + e * FW;
#pragma unroll
        for (int fw = 0; fw < FW; ++fw) {
          acc = fmaf(sxe[pos + fw][e], we[fw], acc);
        }
      }
      outv = tanhf(acc);
    }
    const __bf16 ov = (__bf16)outv;
    Hpad[((size_t)b * LP  + gp) * NFP + f]  = ov;   // [b][pos][f]
    HT  [((size_t)b * NFP + f)  * LP  + gp] = ov;   // [b][f][pos]
  }
}

// =====================================================================
// Kernel 2: fused per-label attention:
//   scores = attn_w x H^T (WMMA bf16) -> LDS
//   softmax over L -> alpha (write d_out + keep in LDS)
//   v = alpha x H (WMMA bf16, B-frags from HT, one v16bf load each)
//   z = sum(fc2_w*v)+b, y_hat=sigmoid(z), BCE partial via atomicAdd
// One block per (batch, 16-label tile); 128 threads = 4 waves.
// =====================================================================
__global__ __launch_bounds__(128)
void caml_attn_fused(const __bf16* __restrict__ Hpad,
                     const __bf16* __restrict__ HT,
                     const float* __restrict__ attn_w,
                     const float* __restrict__ fc2_w,
                     const float* __restrict__ fc2_b,
                     const float* __restrict__ yv,
                     float* __restrict__ out,
                     float* __restrict__ loss_acc) {
  extern __shared__ float smem[];
  float* sc   = smem;                 // [16][LP]  scores, then alpha
  float* vbuf = smem + 16 * LP;       // [4][16][NFP] per-wave partial v
  __shared__ float red[16 * 8];

  const int b      = blockIdx.y;
  const int label0 = blockIdx.x * 16;
  const int tid    = threadIdx.x;
  const int lane   = tid & 31;
  const int wave   = tid >> 5;
  const int m      = lane & 15;       // row (A) / column (B,C) within tile
  const int hi     = lane >> 4;       // lane-half select

  const __bf16* Hb  = Hpad + (size_t)b * LP * NFP;
  const __bf16* HTb = HT   + (size_t)b * NFP * LP;

  // ---- A fragments: attn_w tile [16 labels x 64 (K=f)] as two K=32 frags
  // contiguous K layout: lane<16: halves 0..15 -> K=0..15;
  //                      lane>=16: K=16..31.
  v16bf a0, a1;
  {
    const int gl = label0 + m;
#pragma unroll
    for (int h = 0; h < 16; ++h) {
      const int fo = hi * 16 + h;
      const int f1 = fo + 32;
      float v0 = 0.0f, v1 = 0.0f;
      if (gl < NL) {
        if (fo < NF) v0 = attn_w[(size_t)gl * NF + fo];
        if (f1 < NF) v1 = attn_w[(size_t)gl * NF + f1];
      }
      a0[h] = (__bf16)v0;
      a1[h] = (__bf16)v1;
    }
  }

  // ---- Phase B: scores[16 x LP] = A(16x64) x H^T(64 x LP), tiles of 16 pos
  for (int jt = wave; jt < LP / 16; jt += 4) {
    const int pos = jt * 16 + m;                      // B column = position
    const __bf16* col = Hb + (size_t)pos * NFP;       // 64 contiguous bf16
    if (jt + 4 < LP / 16) {
      __builtin_prefetch(col + 4 * 16 * NFP, 0, 3);   // global_prefetch_b8
    }
    v16bf b0 = *(const v16bf*)(col + hi * 16);        // K = f 0..31 half
    v16bf b1 = *(const v16bf*)(col + 32 + hi * 16);   // K = f 32..63 half
    v8f c = {};
    c = __builtin_amdgcn_wmma_f32_16x16x32_bf16(false, a0, false, b0,
                                                (short)0, c, false, false);
    c = __builtin_amdgcn_wmma_f32_16x16x32_bf16(false, a1, false, b1,
                                                (short)0, c, false, false);
#pragma unroll
    for (int r = 0; r < 8; ++r) {
      sc[(r + 8 * hi) * LP + jt * 16 + m] = c[r];     // C: M=r+8*hi, N=m
    }
  }
  __syncthreads();

  // ---- Phase C: softmax over L per label row (8 threads / row)
  {
    const int row = tid >> 3;
    const int sub = tid & 7;
    float mx = -1e30f;
    for (int p = sub; p < L; p += 8) mx = fmaxf(mx, sc[row * LP + p]);
    red[row * 8 + sub] = mx;
    __syncthreads();
    if (sub == 0) {
      float v = red[row * 8];
#pragma unroll
      for (int i = 1; i < 8; ++i) v = fmaxf(v, red[row * 8 + i]);
      red[row * 8] = v;
    }
    __syncthreads();
    mx = red[row * 8];
    __syncthreads();
    float sum = 0.0f;
    for (int p = sub; p < L; p += 8) {
      const float e = __expf(sc[row * LP + p] - mx);
      sc[row * LP + p] = e;
      sum += e;
    }
    red[row * 8 + sub] = sum;
    __syncthreads();
    if (sub == 0) {
      float v = 0.0f;
#pragma unroll
      for (int i = 0; i < 8; ++i) v += red[row * 8 + i];
      red[row * 8] = v;
    }
    __syncthreads();
    const float inv = 1.0f / red[row * 8];
    const int glr = label0 + row;
    float* arow = out + OUT_ALPHA + ((size_t)b * NL + glr) * (size_t)L;
    for (int p = sub; p < L; p += 8) {
      const float a = sc[row * LP + p] * inv;
      sc[row * LP + p] = a;
      if (glr < NL) arow[p] = a;       // the big (713 MB) output stream
    }
    for (int p = L + sub; p < LP; p += 8) sc[row * LP + p] = 0.0f;  // pad
  }
  __syncthreads();

  // ---- Phase D: v[16 x 64] = alpha(16 x LP) x H(LP x 64), K split on waves
  // B-fragment columns come from HT (feature-major): K=16 contiguous
  // positions -> one 32-byte v16bf load per N-tile per half.
  v8f acc[4] = {v8f{}, v8f{}, v8f{}, v8f{}};
  for (int kt = wave; kt < LP / 32; kt += 4) {
    v16bf aa;
#pragma unroll
    for (int h = 0; h < 16; ++h) {
      const int po = kt * 32 + hi * 16 + h;
      aa[h] = (__bf16)sc[m * LP + po];
    }
    const int k0 = kt * 32 + hi * 16;      // 16 positions (K) in this half
    if (kt + 4 < LP / 32) {
      __builtin_prefetch(HTb + (size_t)m * LP + k0 + 4 * 32, 0, 3);
    }
#pragma unroll
    for (int ft = 0; ft < 4; ++ft) {
      const int f = ft * 16 + m;           // B column = feature
      v16bf bb = *(const v16bf*)(HTb + (size_t)f * LP + k0);
      acc[ft] = __builtin_amdgcn_wmma_f32_16x16x32_bf16(
          false, aa, false, bb, (short)0, acc[ft], false, false);
    }
  }
#pragma unroll
  for (int ft = 0; ft < 4; ++ft) {
#pragma unroll
    for (int r = 0; r < 8; ++r) {
      vbuf[wave * (16 * NFP) + (r + 8 * hi) * NFP + ft * 16 + m] = acc[ft][r];
    }
  }
  __syncthreads();

  // ---- Phase E: z, sigmoid, stable BCE partial sum
  if (tid < 16) {
    const int glr = label0 + tid;
    if (glr < NL) {
      float z = fc2_b[glr];
      for (int f = 0; f < NF; ++f) {
        const float vs = vbuf[0 * (16 * NFP) + tid * NFP + f] +
                         vbuf[1 * (16 * NFP) + tid * NFP + f] +
                         vbuf[2 * (16 * NFP) + tid * NFP + f] +
                         vbuf[3 * (16 * NFP) + tid * NFP + f];
        z = fmaf(fc2_w[(size_t)glr * NF + f], vs, z);
      }
      const float yh = 1.0f / (1.0f + __expf(-z));
      out[OUT_YHAT + (size_t)b * NL + glr] = yh;
      const float yt  = yv[(size_t)b * NL + glr];
      const float t   = log1pf(__expf(-fabsf(z)));
      const float lsp = fminf(z, 0.0f) - t;   // log sigmoid(z)
      const float lsn = fminf(-z, 0.0f) - t;  // log sigmoid(-z)
      atomicAdd(loss_acc, yt * lsp + (1.0f - yt) * lsn);
    }
  }
}

// =====================================================================
// Kernel 3: finalize loss
// =====================================================================
__global__ void caml_loss_finalize(const float* __restrict__ loss_acc,
                                   float* __restrict__ out) {
  out[OUT_LOSS] = -loss_acc[0] / (float)(B * NL);
}

// =====================================================================
extern "C" void kernel_launch(void* const* d_in, const int* in_sizes, int n_in,
                              void* d_out, int out_size, void* d_ws, size_t ws_size,
                              hipStream_t stream) {
  const int*   x      = (const int*)d_in[0];
  const float* yv     = (const float*)d_in[1];
  const float* emb    = (const float*)d_in[2];
  const float* conv_w = (const float*)d_in[3];
  const float* conv_b = (const float*)d_in[4];
  const float* attn_w = (const float*)d_in[5];
  const float* fc2_w  = (const float*)d_in[6];
  const float* fc2_b  = (const float*)d_in[7];
  float* out = (float*)d_out;

  // workspace: Hpad bf16 [B][LP][NFP], HT bf16 [B][NFP][LP], loss float
  const size_t hbytes = (size_t)B * LP * NFP * sizeof(__bf16);
  __bf16* Hpad   = (__bf16*)d_ws;
  __bf16* HT     = (__bf16*)((char*)d_ws + hbytes);
  float* loss_ws = (float*)((char*)d_ws + 2 * hbytes);

  hipMemsetAsync(loss_ws, 0, sizeof(float), stream);

  // Kernel 1: embed + conv + tanh (writes both layouts)
  caml_embed_conv_tanh<<<dim3((LP + 63) / 64, B), 256, 0, stream>>>(
      x, emb, conv_w, conv_b, Hpad, HT);

  // Kernel 2: fused attention (dynamic LDS ~178 KB, OK on 320 KB WGP LDS)
  const size_t smem = (size_t)(16 * LP + 4 * 16 * NFP) * sizeof(float);
  hipFuncSetAttribute((const void*)caml_attn_fused,
                      hipFuncAttributeMaxDynamicSharedMemorySize, (int)smem);
  caml_attn_fused<<<dim3(NLT, B), 128, smem, stream>>>(
      Hpad, HT, attn_w, fc2_w, fc2_b, yv, out, loss_ws);

  // Kernel 3: finalize loss
  caml_loss_finalize<<<1, 1, 0, stream>>>(loss_ws, out);
}
